// Seq2Seq_Attention_80857054315296
// MI455X (gfx1250) — matrix operand order repaired and checked by reading, not compiled, not run on hardware
//
#include <hip/hip_runtime.h>

// Problem constants (fixed by the reference)
#define B_   2048
#define H_   512
#define F_   2
#define L_   20
#define T_   30
#define G4H  2048   // 4*H

typedef __attribute__((ext_vector_type(16))) __bf16       v16bf;
typedef __attribute__((ext_vector_type(8)))  float        v8f;
typedef __attribute__((ext_vector_type(8)))  unsigned int v8u;

__device__ __forceinline__ unsigned short f2bf(float f) {
  unsigned int u = __float_as_uint(f);
  u += 0x7fffu + ((u >> 16) & 1u);   // round-to-nearest-even
  return (unsigned short)(u >> 16);
}

// CDNA5 async global->LDS copy, 16 bytes per lane (tracked by ASYNCcnt)
__device__ __forceinline__ void g2l_b128(unsigned lds_off, const void* gsrc) {
  asm volatile("global_load_async_to_lds_b128 %0, %1, off"
               :: "v"(lds_off), "v"((unsigned long long)(size_t)gsrc)
               : "memory");
}
__device__ __forceinline__ void wait_async_le4() {
  asm volatile("s_wait_asynccnt 4" ::: "memory");
}
__device__ __forceinline__ void wait_async_0() {
  asm volatile("s_wait_asynccnt 0" ::: "memory");
}

// ---------------------------------------------------------------------------
// Weight prep (one-time): all weights stored TRANSPOSED as Wt[K,N] bf16 so the
// GEMM B-tile is a contiguous copy (no per-iteration transpose in the hot loop)
// ---------------------------------------------------------------------------
__global__ void k_cast_bf(unsigned short* __restrict__ dst,
                          const float* __restrict__ src, int n) {
  int i = blockIdx.x * blockDim.x + threadIdx.x;
  if (i < n) dst[i] = f2bf(src[i]);
}

// src [N,K] f32 row-major  ->  dst [K,N] bf16 row-major
__global__ void k_cast_bf_t(unsigned short* __restrict__ dst,
                            const float* __restrict__ src, int N, int K) {
  int i = blockIdx.x * blockDim.x + threadIdx.x;   // over K*N, i = k*N + n
  if (i >= N * K) return;
  int k = i / N, n = i - k * N;
  dst[i] = f2bf(src[(size_t)n * K + k]);
}

// dst [1024, N] = transpose of [wa | wb] where wa,wb are [N,512] f32
__global__ void k_cat_w_t(unsigned short* __restrict__ dst,
                          const float* __restrict__ wa,
                          const float* __restrict__ wb, int N) {
  int i = blockIdx.x * blockDim.x + threadIdx.x;   // over 1024*N, i = k*N + n
  if (i >= 1024 * N) return;
  int k = i / N, n = i - k * N;
  float v = (k < 512) ? wa[(size_t)n * 512 + k] : wb[(size_t)n * 512 + (k - 512)];
  dst[i] = f2bf(v);
}

__global__ void k_add_vec(float* __restrict__ dst, const float* __restrict__ a,
                          const float* __restrict__ b, int n) {
  int i = blockIdx.x * blockDim.x + threadIdx.x;
  if (i < n) dst[i] = a[i] + b[i];
}

// ---------------------------------------------------------------------------
// Per-step elementwise kernels
// ---------------------------------------------------------------------------
// gates[b,j] = bias[j] + x[b,0]*W[j,0] + x[b,1]*W[j,1]   (layer-0 input part, F=2)
__global__ void k_prep_g0(float* __restrict__ gates, const float* __restrict__ xt,
                          const float* __restrict__ wih0, const float* __restrict__ bsum) {
  int i = blockIdx.x * blockDim.x + threadIdx.x;     // over B*4H
  int b = i >> 11, j = i & 2047;
  gates[i] = bsum[j] + xt[b * 2] * wih0[j * 2] + xt[b * 2 + 1] * wih0[j * 2 + 1];
}

__global__ void k_fill_rows(float* __restrict__ dst, const float* __restrict__ bias,
                            int mask, int total) {
  int i = blockIdx.x * blockDim.x + threadIdx.x;
  if (i < total) dst[i] = bias[i & mask];
}

// dst [B, 2H] bf16 = concat(s0, s1) along hidden dim
__global__ void k_cast_concat(unsigned short* __restrict__ dst,
                              const float* __restrict__ s0,
                              const float* __restrict__ s1) {
  int i = blockIdx.x * blockDim.x + threadIdx.x;     // over B*1024
  int b = i >> 10, k = i & 1023;
  float v = (k < H_) ? s0[b * H_ + k] : s1[b * H_ + (k - H_)];
  dst[i] = f2bf(v);
}

// LSTM gate activation (PyTorch order i,f,g,o)
__global__ void k_lstm_act(float* __restrict__ h, float* __restrict__ c,
                           const float* __restrict__ gates) {
  int i = blockIdx.x * blockDim.x + threadIdx.x;     // over B*H
  int b = i >> 9, j = i & 511;
  const float* g = gates + (size_t)b * G4H;
  float gi = g[j], gf = g[H_ + j], gg = g[2 * H_ + j], go = g[3 * H_ + j];
  float si = 1.f / (1.f + __expf(-gi));
  float sf = 1.f / (1.f + __expf(-gf));
  float so = 1.f / (1.f + __expf(-go));
  float cn = sf * c[i] + si * tanhf(gg);
  c[i] = cn;
  h[i] = so * tanhf(cn);
}

// row-wise softmax over 512 cols, in place; one 256-thread block per row
__global__ __launch_bounds__(256) void k_softmax512(float* __restrict__ w) {
  __shared__ float red[256];
  int row = blockIdx.x, t = threadIdx.x;
  float* p = w + (size_t)row * 512;
  float v0 = p[t], v1 = p[t + 256];
  red[t] = fmaxf(v0, v1);
  __syncthreads();
  for (int s = 128; s > 0; s >>= 1) {
    if (t < s) red[t] = fmaxf(red[t], red[t + s]);
    __syncthreads();
  }
  float m = red[0];
  __syncthreads();
  float e0 = __expf(v0 - m), e1 = __expf(v1 - m);
  red[t] = e0 + e1;
  __syncthreads();
  for (int s = 128; s > 0; s >>= 1) {
    if (t < s) red[t] += red[t + s];
    __syncthreads();
  }
  float inv = 1.f / red[0];
  p[t] = e0 * inv;
  p[t + 256] = e1 * inv;
}

// encoder combine: scale = sum(h)/sum(h*w); top = h*w*scale; h = top; outhid = top
__global__ __launch_bounds__(256) void k_enc_comb(float* __restrict__ h,
                                                  const float* __restrict__ w,
                                                  float* __restrict__ outhid) {
  __shared__ float r1[256], r2[256];
  int row = blockIdx.x, t = threadIdx.x;
  float* hp = h + (size_t)row * 512;
  const float* wp = w + (size_t)row * 512;
  float a0 = hp[t], a1 = hp[t + 256], w0 = wp[t], w1 = wp[t + 256];
  r1[t] = a0 + a1;
  r2[t] = a0 * w0 + a1 * w1;
  __syncthreads();
  for (int s = 128; s > 0; s >>= 1) {
    if (t < s) { r1[t] += r1[t + s]; r2[t] += r2[t + s]; }
    __syncthreads();
  }
  float scale = r1[0] / r2[0];
  float t0 = a0 * w0 * scale, t1 = a1 * w1 * scale;
  hp[t] = t0; hp[t + 256] = t1;
  float* op = outhid + (size_t)row * 512;
  op[t] = t0; op[t + 256] = t1;
}

// decoder combine: top = h*w; h = top; outhid = top
__global__ void k_dec_comb(float* __restrict__ h, const float* __restrict__ w,
                           float* __restrict__ outhid) {
  int i = blockIdx.x * blockDim.x + threadIdx.x;     // over B*H
  float v = h[i] * w[i];
  h[i] = v;
  outhid[i] = v;
}

// decoder projection: out[b,f] = h[b,:] . out_W[f,:] + out_b[f]   (F=2)
__global__ void k_pred(float* __restrict__ out, const float* __restrict__ h,
                       const float* __restrict__ outW, const float* __restrict__ outb) {
  int i = blockIdx.x * blockDim.x + threadIdx.x;     // over B*2
  if (i >= B_ * F_) return;
  int b = i >> 1, f = i & 1;
  const float* hp = h + (size_t)b * H_;
  const float* wp = outW + f * H_;
  float acc = 0.f;
  #pragma unroll 8
  for (int k = 0; k < H_; ++k) acc += hp[k] * wp[k];
  out[i] = acc + outb[f];
}

// ---------------------------------------------------------------------------
// WMMA GEMM: C[M,N] += A[M,K] * Wt[K,N]   (A,Wt bf16 row-major, C f32)
// block = 256 threads (8 wave32), block tile 128x128, K-step 32.
// Double-buffered LDS fed by GLOBAL_LOAD_ASYNC_TO_LDS_B128 (ASYNCcnt).
// Each wave computes a 32(M)x64(N) tile = 2x4 accumulators of v8f.
// ---------------------------------------------------------------------------
__global__ __launch_bounds__(256)
void k_gemm_bf16(float* __restrict__ C, const unsigned short* __restrict__ A,
                 const unsigned short* __restrict__ Wt, int N, int K) {
  __shared__ unsigned short sA[2][128][32];    // A tiles   (2 x 8KB)
  __shared__ unsigned short sB[2][32][128];    // Wt tiles  (2 x 8KB), sB[k][n]

  const int n0 = blockIdx.x * 128;
  const int m0 = blockIdx.y * 128;
  const int tid  = threadIdx.x;
  const int lane = tid & 31;
  const int wave = tid >> 5;
  const int wm = (wave & 3) * 32;              // wave M offset in block tile
  const int wn = (wave >> 2) * 64;             // wave N offset in block tile
  const int hi = lane >> 4;                    // lane half (A layout / C rows)
  const int lr = lane & 15;

  // per-thread staging chunks: 2x16B for A tile (128x32), 2x16B for B tile (32x128)
  const int rA0 = tid >> 2,          cA0 = (tid & 3) * 8;
  const int rA1 = (tid + 256) >> 2,  cA1 = ((tid + 256) & 3) * 8;
  const int rB0 = tid >> 4,          cB0 = (tid & 15) * 8;
  const int rB1 = (tid + 256) >> 4,  cB1 = ((tid + 256) & 15) * 8;

  auto issue_tiles = [&](int buf, int k0) {
    g2l_b128((unsigned)(size_t)(const void*)&sA[buf][rA0][cA0],
             A + (size_t)(m0 + rA0) * K + (k0 + cA0));
    g2l_b128((unsigned)(size_t)(const void*)&sA[buf][rA1][cA1],
             A + (size_t)(m0 + rA1) * K + (k0 + cA1));
    g2l_b128((unsigned)(size_t)(const void*)&sB[buf][rB0][cB0],
             Wt + (size_t)(k0 + rB0) * N + (n0 + cB0));
    g2l_b128((unsigned)(size_t)(const void*)&sB[buf][rB1][cB1],
             Wt + (size_t)(k0 + rB1) * N + (n0 + cB1));
  };

  v8f acc[2][4];
  const v8f vzero = {0.f, 0.f, 0.f, 0.f, 0.f, 0.f, 0.f, 0.f};
  #pragma unroll
  for (int mi = 0; mi < 2; ++mi)
    #pragma unroll
    for (int ni = 0; ni < 4; ++ni) acc[mi][ni] = vzero;

  issue_tiles(0, 0);
  const int nk = K >> 5;
  for (int i = 0; i < nk; ++i) {
    const int cur = i & 1;
    if (i + 1 < nk) {
      issue_tiles(cur ^ 1, (i + 1) << 5);   // prefetch next tiles into other buffer
      wait_async_le4();                     // only the 4 newest (next-tile) may remain
    } else {
      wait_async_0();
    }
    __syncthreads();

    // A 16x32 bf16 fragment: lane row m=lr; halves0-7 -> K=8*hi.., halves8-15 -> K=16+8*hi..
    v16bf afrag[2];
    #pragma unroll
    for (int mi = 0; mi < 2; ++mi) {
      const unsigned int* p = (const unsigned int*)&sA[cur][wm + mi * 16 + lr][0];
      v8u t;
      #pragma unroll
      for (int q = 0; q < 4; ++q) t[q] = p[4 * hi + q];
      #pragma unroll
      for (int q = 0; q < 4; ++q) t[4 + q] = p[8 + 4 * hi + q];
      afrag[mi] = __builtin_bit_cast(v16bf, t);
    }
    // B 32x16 fragment: lane col n=lr; halves0-7 -> K=8*hi.., halves8-15 -> K=16+8*hi..
    v16bf bfrag[4];
    #pragma unroll
    for (int ni = 0; ni < 4; ++ni) {
      const unsigned short* p = &sB[cur][0][wn + ni * 16 + lr];
      v8u t;
      #pragma unroll
      for (int q = 0; q < 4; ++q)
        t[q] = (unsigned)p[(8 * hi + 2 * q) * 128] | ((unsigned)p[(8 * hi + 2 * q + 1) * 128] << 16);
      #pragma unroll
      for (int q = 0; q < 4; ++q)
        t[4 + q] = (unsigned)p[(16 + 8 * hi + 2 * q) * 128] | ((unsigned)p[(16 + 8 * hi + 2 * q + 1) * 128] << 16);
      bfrag[ni] = __builtin_bit_cast(v16bf, t);
    }

    #pragma unroll
    for (int mi = 0; mi < 2; ++mi)
      #pragma unroll
      for (int ni = 0; ni < 4; ++ni)
        acc[mi][ni] = __builtin_amdgcn_wmma_f32_16x16x32_bf16(
            false, afrag[mi], false, bfrag[ni],
            (short)0, acc[mi][ni], false, false);

    __syncthreads();
  }

  // epilogue: C += acc (C/D layout: lane -> col, reg + lane-half -> row)
  #pragma unroll
  for (int mi = 0; mi < 2; ++mi) {
    #pragma unroll
    for (int ni = 0; ni < 4; ++ni) {
      int n = n0 + wn + ni * 16 + lr;
      int mbase = m0 + wm + mi * 16 + 8 * hi;
      #pragma unroll
      for (int r = 0; r < 8; ++r) {
        float* p = C + (size_t)(mbase + r) * N + n;
        *p = *p + acc[mi][ni][r];
      }
    }
  }
}

static inline void launch_gemm(float* C, const unsigned short* A, const unsigned short* Wt,
                               int M, int N, int K, hipStream_t s) {
  dim3 grid(N / 128, M / 128), block(256);
  k_gemm_bf16<<<grid, block, 0, s>>>(C, A, Wt, N, K);
}

// ---------------------------------------------------------------------------
// Host orchestration
// ---------------------------------------------------------------------------
extern "C" void kernel_launch(void* const* d_in, const int* in_sizes, int n_in,
                              void* d_out, int out_size, void* d_ws, size_t ws_size,
                              hipStream_t stream) {
  (void)in_sizes; (void)n_in; (void)out_size; (void)ws_size;

  const float* x     = (const float*)d_in[0];   // [20,B,2]
  const float* y     = (const float*)d_in[1];   // [30,B,2]
  const float* hid   = (const float*)d_in[2];   // [50,B,H]
  const float* h0i   = (const float*)d_in[3];   // [2,B,H]
  const float* c0i   = (const float*)d_in[4];   // [2,B,H]
  const float* attW  = (const float*)d_in[5];   // [H,2H]
  const float* attb  = (const float*)d_in[6];   // [H]
  const float* eWih0 = (const float*)d_in[7];
  const float* eWhh0 = (const float*)d_in[8];
  const float* ebih0 = (const float*)d_in[9];
  const float* ebhh0 = (const float*)d_in[10];
  const float* eWih1 = (const float*)d_in[11];
  const float* eWhh1 = (const float*)d_in[12];
  const float* ebih1 = (const float*)d_in[13];
  const float* ebhh1 = (const float*)d_in[14];
  const float* dWih0 = (const float*)d_in[15];
  const float* dWhh0 = (const float*)d_in[16];
  const float* dbih0 = (const float*)d_in[17];
  const float* dbhh0 = (const float*)d_in[18];
  const float* dWih1 = (const float*)d_in[19];
  const float* dWhh1 = (const float*)d_in[20];
  const float* dbih1 = (const float*)d_in[21];
  const float* dbhh1 = (const float*)d_in[22];
  const float* outW  = (const float*)d_in[23];  // [2,H]
  const float* outb  = (const float*)d_in[24];  // [2]

  // ---- workspace layout ----
  char* wp = (char*)d_ws;
  auto alloc = [&](size_t bytes) -> void* {
    void* r = (void*)wp;
    wp += (bytes + 255) & ~(size_t)255;
    return r;
  };
  // transposed bf16 weights: Wt[K,N]
  unsigned short* eWhh0_t = (unsigned short*)alloc((size_t)512 * 2048 * 2);
  unsigned short* eW1_t   = (unsigned short*)alloc((size_t)1024 * 2048 * 2);
  unsigned short* dWhh0_t = (unsigned short*)alloc((size_t)512 * 2048 * 2);
  unsigned short* dW1_t   = (unsigned short*)alloc((size_t)1024 * 2048 * 2);
  unsigned short* attW_t  = (unsigned short*)alloc((size_t)1024 * 512 * 2);
  float* eB0 = (float*)alloc(2048 * 4);
  float* eB1 = (float*)alloc(2048 * 4);
  float* dB0 = (float*)alloc(2048 * 4);
  float* dB1 = (float*)alloc(2048 * 4);
  float* h0s = (float*)alloc((size_t)B_ * H_ * 4);
  float* h1s = (float*)alloc((size_t)B_ * H_ * 4);
  float* c0s = (float*)alloc((size_t)B_ * H_ * 4);
  float* c1s = (float*)alloc((size_t)B_ * H_ * 4);
  float* gates = (float*)alloc((size_t)B_ * G4H * 4);
  unsigned short* abuf = (unsigned short*)alloc((size_t)B_ * 1024 * 2);
  float* attlog = (float*)alloc((size_t)B_ * 512 * 4);

  float* outputs = (float*)d_out;                              // [T,B,2]
  float* outhid  = (float*)d_out + (size_t)T_ * B_ * F_;       // [L+T,B,H]

  const int TB = 256;

  // ---- one-time weight prep (re-run every call for determinism) ----
  k_cast_bf_t<<<(512 * 2048) / TB, TB, 0, stream>>>(eWhh0_t, eWhh0, 2048, 512);
  k_cast_bf_t<<<(512 * 2048) / TB, TB, 0, stream>>>(dWhh0_t, dWhh0, 2048, 512);
  k_cast_bf_t<<<(1024 * 512) / TB, TB, 0, stream>>>(attW_t, attW, 512, 1024);
  k_cat_w_t<<<(1024 * 2048) / TB, TB, 0, stream>>>(eW1_t, eWih1, eWhh1, 2048);
  k_cat_w_t<<<(1024 * 2048) / TB, TB, 0, stream>>>(dW1_t, dWih1, dWhh1, 2048);
  k_add_vec<<<2048 / TB, TB, 0, stream>>>(eB0, ebih0, ebhh0, 2048);
  k_add_vec<<<2048 / TB, TB, 0, stream>>>(eB1, ebih1, ebhh1, 2048);
  k_add_vec<<<2048 / TB, TB, 0, stream>>>(dB0, dbih0, dbhh0, 2048);
  k_add_vec<<<2048 / TB, TB, 0, stream>>>(dB1, dbih1, dbhh1, 2048);

  // ---- init recurrent state from h0/c0 ----
  const size_t SBH = (size_t)B_ * H_ * sizeof(float);
  hipMemcpyAsync(h0s, h0i, SBH, hipMemcpyDeviceToDevice, stream);
  hipMemcpyAsync(h1s, h0i + (size_t)B_ * H_, SBH, hipMemcpyDeviceToDevice, stream);
  hipMemcpyAsync(c0s, c0i, SBH, hipMemcpyDeviceToDevice, stream);
  hipMemcpyAsync(c1s, c0i + (size_t)B_ * H_, SBH, hipMemcpyDeviceToDevice, stream);

  const int nBH  = (B_ * H_) / TB;
  const int nB4H = (B_ * G4H) / TB;
  const int nB2H = (B_ * 1024) / TB;
  const int nBHa = (B_ * 512) / TB;

  // ================= encoder =================
  for (int t = 0; t < L_; ++t) {
    // layer 0
    k_prep_g0<<<nB4H, TB, 0, stream>>>(gates, x + (size_t)t * B_ * F_, eWih0, eB0);
    k_cast_bf<<<nBH, TB, 0, stream>>>(abuf, h0s, B_ * H_);
    launch_gemm(gates, abuf, eWhh0_t, B_, G4H, 512, stream);
    k_lstm_act<<<nBH, TB, 0, stream>>>(h0s, c0s, gates);
    // layer 1 (fused input+recurrent GEMM, K=1024)
    k_fill_rows<<<nB4H, TB, 0, stream>>>(gates, eB1, 2047, B_ * G4H);
    k_cast_concat<<<nB2H, TB, 0, stream>>>(abuf, h0s, h1s);
    launch_gemm(gates, abuf, eW1_t, B_, G4H, 1024, stream);
    k_lstm_act<<<nBH, TB, 0, stream>>>(h1s, c1s, gates);
    // attention
    k_cast_concat<<<nB2H, TB, 0, stream>>>(abuf, h1s, hid + (size_t)t * B_ * H_);
    k_fill_rows<<<nBHa, TB, 0, stream>>>(attlog, attb, 511, B_ * 512);
    launch_gemm(attlog, abuf, attW_t, B_, 512, 1024, stream);
    k_softmax512<<<B_, TB, 0, stream>>>(attlog);
    k_enc_comb<<<B_, TB, 0, stream>>>(h1s, attlog, outhid + (size_t)t * B_ * H_);
  }

  // ================= decoder =================
  for (int t = 0; t < T_; ++t) {
    const float* inp = (t == 0) ? (x + (size_t)(L_ - 1) * B_ * F_)
                                : (y + (size_t)(t - 1) * B_ * F_);
    // layer 0
    k_prep_g0<<<nB4H, TB, 0, stream>>>(gates, inp, dWih0, dB0);
    k_cast_bf<<<nBH, TB, 0, stream>>>(abuf, h0s, B_ * H_);
    launch_gemm(gates, abuf, dWhh0_t, B_, G4H, 512, stream);
    k_lstm_act<<<nBH, TB, 0, stream>>>(h0s, c0s, gates);
    // layer 1
    k_fill_rows<<<nB4H, TB, 0, stream>>>(gates, dB1, 2047, B_ * G4H);
    k_cast_concat<<<nB2H, TB, 0, stream>>>(abuf, h0s, h1s);
    launch_gemm(gates, abuf, dW1_t, B_, G4H, 1024, stream);
    k_lstm_act<<<nBH, TB, 0, stream>>>(h1s, c1s, gates);
    // output projection (from pre-attention h1)
    k_pred<<<(B_ * F_ + TB - 1) / TB, TB, 0, stream>>>(
        outputs + (size_t)t * B_ * F_, h1s, outW, outb);
    // attention
    k_cast_concat<<<nB2H, TB, 0, stream>>>(abuf, h1s, hid + (size_t)(3 + t) * B_ * H_);
    k_fill_rows<<<nBHa, TB, 0, stream>>>(attlog, attb, 511, B_ * 512);
    launch_gemm(attlog, abuf, attW_t, B_, 512, 1024, stream);
    k_softmax512<<<B_, TB, 0, stream>>>(attlog);
    k_dec_comb<<<nBH, TB, 0, stream>>>(h1s, attlog, outhid + (size_t)(L_ + t) * B_ * H_);
  }
}
